// MoGIN2_86225763434550
// MI455X (gfx1250) — matrix operand order repaired and checked by reading, not compiled, not run on hardware
//
#include <hip/hip_runtime.h>
#include <math.h>

typedef __attribute__((ext_vector_type(16))) _Float16 v16h;
typedef __attribute__((ext_vector_type(8)))  _Float16 v8h;
typedef __attribute__((ext_vector_type(2)))  _Float16 h2;
typedef __attribute__((ext_vector_type(8)))  float    v8f;

union V16U { v16h v; v8h h[2]; };

// ---------------- problem constants ----------------
static constexpr int    NG   = 32;              // graphs
static constexpr int    NPG  = 512;             // nodes per graph
static constexpr int    NN   = NG * NPG;        // 16384 nodes
static constexpr int    DEG  = 32;
static constexpr long   EE   = (long)NN * DEG;  // 524288 edges
static constexpr int    DD   = 128;
static constexpr int    HH   = 256;
static constexpr int    NE   = 8;
static constexpr int    EHID = 64;
static constexpr int    LL   = 3;
static constexpr float  CUTOFF = 10.0f;

// ---------------- workspace layout ----------------
static constexpr size_t alignup(size_t x) { return (x + 255) & ~(size_t)255; }
static constexpr size_t OFF_DIST  = 0;                                         // E f32
static constexpr size_t OFF_ZERO  = alignup(OFF_DIST  + EE * 4);               // easum(24)+gsum(32)+gsum2(32)
static constexpr size_t OFF_GMEAN = alignup(OFF_ZERO  + 128 * 4);              // 32 f32
static constexpr size_t OFF_GISTD = alignup(OFF_GMEAN + 32 * 4);               // 32 f32
static constexpr size_t OFF_B2SUM = alignup(OFF_GISTD + 32 * 4);               // L*128 f32
static constexpr size_t OFF_SUMX  = alignup(OFF_B2SUM + LL * 128 * 4);         // 32*128 f32
static constexpr size_t OFF_SUMX2 = alignup(OFF_SUMX  + NG * DD * 4);          // 32*128 f32
static constexpr size_t OFF_EAW   = alignup(OFF_SUMX2 + NG * DD * 4);          // E*8 f32
static constexpr size_t OFF_AGG   = alignup(OFF_EAW   + EE * NE * 4);          // 8*N*128 f16
static constexpr size_t OFF_HA    = alignup(OFF_AGG   + (size_t)NE * NN * DD * 2); // N*128 f16
static constexpr size_t OFF_HPRE  = alignup(OFF_HA    + (size_t)NN * DD * 2);  // N*128 f32
static constexpr size_t OFF_W1T   = alignup(OFF_HPRE  + (size_t)NN * DD * 4);  // L*8*256*128 f16
static constexpr size_t OFF_W2T   = alignup(OFF_W1T   + (size_t)LL * NE * HH * DD * 2);

// ---------------- kernels ----------------

// h0[i,c] = emb[atom_type[i], c]  (f16 node features)
__global__ __launch_bounds__(256) void init_h_kernel(const float* __restrict__ emb,
                                                     const int* __restrict__ atype,
                                                     _Float16* __restrict__ h) {
  size_t i = (size_t)blockIdx.x * 256 + threadIdx.x;  // N*128 total
  int node = (int)(i >> 7);
  int c = (int)(i & 127);
  h[i] = (_Float16)emb[(size_t)atype[node] * DD + c];
}

// per-edge distance + per-graph sum(d), sum(d^2)
__global__ __launch_bounds__(256) void dist_kernel(const float* __restrict__ pos,
                                                   const int* __restrict__ esrc,
                                                   float* __restrict__ dist,
                                                   float* __restrict__ gsum,
                                                   float* __restrict__ gsum2) {
  __shared__ float s1[256], s2[256];
  int t = threadIdx.x;
  size_t e = (size_t)blockIdx.x * 256 + t;
  int s = esrc[e];
  int dn = (int)(e >> 5);  // dst = e / DEG (edges sorted by dst by construction)
  float dx = pos[s * 3 + 0] - pos[dn * 3 + 0];
  float dy = pos[s * 3 + 1] - pos[dn * 3 + 1];
  float dz = pos[s * 3 + 2] - pos[dn * 3 + 2];
  float dd = sqrtf(dx * dx + dy * dy + dz * dz);
  dist[e] = dd;
  s1[t] = dd; s2[t] = dd * dd;
  __syncthreads();
  for (int o = 128; o > 0; o >>= 1) {
    if (t < o) { s1[t] += s1[t + o]; s2[t] += s2[t + o]; }
    __syncthreads();
  }
  if (t == 0) {
    int g = (int)(((size_t)blockIdx.x * 256) >> 14);  // 16384 edges per graph
    atomicAdd(&gsum[g], s1[0]);
    atomicAdd(&gsum2[g], s2[0]);
  }
}

// per-graph mean & inv-std of edge distances (GraphNorm with mean-scale ms)
__global__ void gstat_kernel(const float* __restrict__ gsum, const float* __restrict__ gsum2,
                             const float* __restrict__ dnms,
                             float* __restrict__ gmean, float* __restrict__ gistd) {
  int g = threadIdx.x;
  if (g >= NG) return;
  const float inv = 1.0f / 16384.0f;
  float mean = gsum[g] * inv;
  float ex2 = gsum2[g] * inv;
  float ms = dnms[0];
  float m = ms * mean;
  float var = ex2 - 2.0f * m * mean + m * m;
  gmean[g] = mean;
  gistd[g] = rsqrtf(var + 1e-5f);
}

// pre-transpose expert weights to [out][in] f16 for contiguous WMMA B fragments
__global__ __launch_bounds__(256) void convert_w_kernel(const float* __restrict__ nw1,
                                                        const float* __restrict__ nw2,
                                                        _Float16* __restrict__ w1t,
                                                        _Float16* __restrict__ w2t) {
  int i = blockIdx.x * 256 + threadIdx.x;  // 0 .. L*8*128*256-1
  int le = i >> 15;                        // l*8+e
  int rem = i & 32767;
  int n1 = rem >> 7, k1 = rem & 127;       // w1t[le][n(256)][k(128)]
  w1t[i] = (_Float16)nw1[(size_t)le * 32768 + (size_t)k1 * HH + n1];
  int n2 = rem >> 8, k2 = rem & 255;       // w2t[le][n(128)][k(256)]
  w2t[i] = (_Float16)nw2[(size_t)le * 32768 + (size_t)k2 * DD + n2];
}

// b2sum[l][c] = sum_e node_b2[l][e][c]
__global__ void b2sum_kernel(const float* __restrict__ nb2, float* __restrict__ b2s) {
  int i = blockIdx.x * 128 + threadIdx.x;
  if (i >= LL * DD) return;
  int l = i >> 7, c = i & 127;
  float s = 0.f;
  for (int e = 0; e < NE; e++) s += nb2[((size_t)(l * NE + e) << 7) + c];
  b2s[i] = s;
}

// edge router: scalar GraphNorm feature -> ReLU(64) -> softmax(8); stores eaw, accumulates sum(ea)
__global__ __launch_bounds__(256) void router_kernel(const float* __restrict__ dist,
                                                     const float* __restrict__ gmean,
                                                     const float* __restrict__ gistd,
                                                     const float* __restrict__ ew1,
                                                     const float* __restrict__ eb1,
                                                     const float* __restrict__ ew2,
                                                     const float* __restrict__ eb2,
                                                     const float* __restrict__ dnw,
                                                     const float* __restrict__ dnb,
                                                     const float* __restrict__ dnms,
                                                     float* __restrict__ eaw,
                                                     float* __restrict__ easum) {
  __shared__ float s_w1[EHID], s_b1[EHID], s_w2[EHID * NE], s_b2[NE], s_red[NE];
  int t = threadIdx.x;
  if (t < EHID) { s_w1[t] = ew1[t]; s_b1[t] = eb1[t]; }
  if (t < NE) { s_b2[t] = eb2[t]; s_red[t] = 0.f; }
  for (int i = t; i < EHID * NE; i += 256) s_w2[i] = ew2[i];
  __syncthreads();

  size_t e = (size_t)blockIdx.x * 256 + t;
  float d = dist[e];
  int g = (int)(e >> 14);
  float ea0 = dnw[0] * (d - dnms[0] * gmean[g]) * gistd[g] + dnb[0];

  float logit[NE];
#pragma unroll
  for (int j = 0; j < NE; j++) logit[j] = s_b2[j];
  for (int k = 0; k < EHID; k++) {
    float hv = fmaxf(ea0 * s_w1[k] + s_b1[k], 0.0f);
#pragma unroll
    for (int j = 0; j < NE; j++) logit[j] += hv * s_w2[k * NE + j];
  }
  float mx = logit[0];
#pragma unroll
  for (int j = 1; j < NE; j++) mx = fmaxf(mx, logit[j]);
  float ex[NE], se = 0.f;
#pragma unroll
  for (int j = 0; j < NE; j++) { ex[j] = expf(logit[j] - mx); se += ex[j]; }
  float inv = 1.0f / se;
  float ewgt = (CUTOFF - d) * (1.0f / CUTOFF);
#pragma unroll
  for (int j = 0; j < NE; j++) {
    float ea = ex[j] * inv;
    eaw[e * NE + j] = ea * ewgt;
    atomicAdd(&s_red[j], ea);
  }
  __syncthreads();
  if (t < NE) atomicAdd(&easum[t], s_red[t]);
}

// per-node aggregation over 32 contiguous edges; thread = (node, channel-pair);
// one b32 load per edge (2 f16 channels), 8 expert f32 accumulators each, packed b32 stores.
__global__ __launch_bounds__(256) void aggregate_kernel(const _Float16* __restrict__ h,
                                                        const int* __restrict__ esrc,
                                                        const float* __restrict__ eaw,
                                                        _Float16* __restrict__ agg) {
  __shared__ int s_src[128];
  __shared__ float s_eaw[1024];
  int t = threadIdx.x;
  size_t eb = (size_t)blockIdx.x * 128;  // 4 nodes * 32 edges
  if (t < 128) s_src[t] = esrc[eb + t];
#pragma unroll
  for (int i = 0; i < 4; i++) s_eaw[i * 256 + t] = eaw[eb * NE + i * 256 + t];
  __syncthreads();

  int nl = t >> 6;      // node 0..3 within block
  int cp = t & 63;      // channel pair 0..63
  float acc0[NE], acc1[NE];
#pragma unroll
  for (int j = 0; j < NE; j++) { acc0[j] = 0.f; acc1[j] = 0.f; }
#pragma unroll 4
  for (int k = 0; k < DEG; k++) {
    int idx = nl * 32 + k;
    h2 x = *(const h2*)(h + (size_t)s_src[idx] * DD + cp * 2);
    float x0 = (float)x[0], x1 = (float)x[1];
    const float* wv = &s_eaw[idx * NE];
#pragma unroll
    for (int j = 0; j < NE; j++) { acc0[j] += wv[j] * x0; acc1[j] += wv[j] * x1; }
  }
  size_t node = (size_t)blockIdx.x * 4 + nl;
#pragma unroll
  for (int j = 0; j < NE; j++) {
    h2 o;
    o[0] = (_Float16)acc0[j];
    o[1] = (_Float16)acc1[j];
    *(h2*)(agg + ((size_t)j * NN + node) * DD + cp * 2) = o;
  }
}

// MoE expert MLPs with WMMA f16->f32: block = 64 nodes, 8 waves
__global__ __launch_bounds__(256) void moe_gemm_kernel(const _Float16* __restrict__ agg,  // [8][N][128]
                                                       const _Float16* __restrict__ w1t,  // [8][256][128]
                                                       const _Float16* __restrict__ w2t,  // [8][128][256]
                                                       const float* __restrict__ b1,      // [8][256]
                                                       const float* __restrict__ b2s,     // [128]
                                                       float* __restrict__ out) {         // [N][128]
  __shared__ _Float16 s_mid[64][264];  // relu(GEMM1) tile, padded rows (528B ≡ 4 banks mod 64)
  const int t = threadIdx.x;
  const int lane = t & 31;
  const int w = t >> 5;    // wave 0..7
  const int mt = w & 3;    // M-tile of 16 rows
  const int hw = w >> 2;   // N half
  const int lm = lane & 15;
  const int lh = lane >> 4;
  const size_t m0 = (size_t)blockIdx.x * 64 + (size_t)mt * 16;

  v8f dacc[4];
#pragma unroll
  for (int i = 0; i < 4; i++)
#pragma unroll
    for (int r = 0; r < 8; r++) dacc[i][r] = 0.f;

  for (int e = 0; e < NE; ++e) {
    // ---- GEMM1: C1 = relu(agg_e @ W1_e + b1_e), rows [m0,m0+16), cols [hw*128, +128) ----
    V16U a1[4];
    const _Float16* arow = agg + ((size_t)e * NN + m0 + lm) * DD;
#pragma unroll
    for (int kk = 0; kk < 4; kk++) {  // A layout matching B: K = kb+16h+0..15
      a1[kk].h[0] = *(const v8h*)(arow + kk * 32 + 16 * lh);
      a1[kk].h[1] = *(const v8h*)(arow + kk * 32 + 16 * lh + 8);
    }
#pragma unroll
    for (int nt = 0; nt < 8; ++nt) {
      int ncol = hw * 128 + nt * 16 + lm;
      float bias = b1[e * HH + ncol];
      v8f c = {};  // zero accumulator -> first WMMA can use inline-0 SRC2; bias folded in epilogue
      const _Float16* brow = w1t + ((size_t)e * HH + ncol) * DD;
#pragma unroll
      for (int kk = 0; kk < 4; kk++) {  // B fragment: 16 consecutive halves at K = kb+16h
        v16h b = *(const v16h*)(brow + kk * 32 + 16 * lh);
        c = __builtin_amdgcn_wmma_f32_16x16x32_f16(false, a1[kk].v, false, b, (short)0, c, false, false);
      }
#pragma unroll
      for (int r = 0; r < 8; r++) {  // C/D: M = r + 8*lh, N = lm
        float v = c[r] + bias;
        s_mid[mt * 16 + 8 * lh + r][hw * 128 + nt * 16 + lm] = (_Float16)(v > 0.f ? v : 0.f);
      }
    }
    __syncthreads();
    // ---- GEMM2: out += relu(C1) @ W2_e, cols [hw*64, +64) ----
#pragma unroll
    for (int kk = 0; kk < 8; kk++) {
      V16U a2;
      a2.h[0] = *(const v8h*)(&s_mid[mt * 16 + lm][kk * 32 + 16 * lh]);
      a2.h[1] = *(const v8h*)(&s_mid[mt * 16 + lm][kk * 32 + 16 * lh + 8]);
#pragma unroll
      for (int nt = 0; nt < 4; ++nt) {
        int ncol = hw * 64 + nt * 16 + lm;
        const _Float16* brow = w2t + ((size_t)e * DD + ncol) * HH;
        v16h b = *(const v16h*)(brow + kk * 32 + 16 * lh);
        dacc[nt] = __builtin_amdgcn_wmma_f32_16x16x32_f16(false, a2.v, false, b, (short)0, dacc[nt], false, false);
      }
    }
    __syncthreads();
  }
#pragma unroll
  for (int nt = 0; nt < 4; nt++) {
    int ncol = hw * 64 + nt * 16 + lm;
    float bb = b2s[ncol];
#pragma unroll
    for (int r = 0; r < 8; r++)
      out[(m0 + 8 * lh + r) * DD + ncol] = dacc[nt][r] + bb;
  }
}

// per-graph, per-channel sum(x), sum(x^2)
__global__ __launch_bounds__(256) void norm_reduce_kernel(const float* __restrict__ x,
                                                          float* __restrict__ sumx,
                                                          float* __restrict__ sumx2) {
  __shared__ float s1[256], s2[256];
  int g = blockIdx.x, t = threadIdx.x;
  int c = t & 127, part = t >> 7;
  float a = 0.f, a2 = 0.f;
  for (int n = part; n < NPG; n += 2) {
    float v = x[((size_t)g * NPG + n) * DD + c];
    a += v; a2 += v * v;
  }
  s1[t] = a; s2[t] = a2;
  __syncthreads();
  if (part == 0) {
    sumx[g * DD + c] = s1[t] + s1[t + 128];
    sumx2[g * DD + c] = s2[t] + s2[t + 128];
  }
}

// h = tanh(GraphNorm(x))  (written as f16 node features for the next layer's gather)
__global__ __launch_bounds__(256) void norm_apply_kernel(const float* __restrict__ x,
                                                         const float* __restrict__ sumx,
                                                         const float* __restrict__ sumx2,
                                                         const float* __restrict__ gw,
                                                         const float* __restrict__ gb,
                                                         const float* __restrict__ gms,
                                                         _Float16* __restrict__ h) {
  size_t i = (size_t)blockIdx.x * 256 + threadIdx.x;
  int c = (int)(i & 127);
  int g = (int)(i >> 16);  // 512*128 elements per graph
  const float inv = 1.0f / (float)NPG;
  float mean = sumx[g * DD + c] * inv;
  float ex2 = sumx2[g * DD + c] * inv;
  float ms = gms[c];
  float m = ms * mean;
  float var = ex2 - 2.0f * m * mean + m * m;
  float v = (x[i] - m) * rsqrtf(var + 1e-5f);
  h[i] = (_Float16)tanhf(gw[c] * v + gb[c]);
}

// pooled[g] = mean over (nodes, channels)
__global__ __launch_bounds__(256) void pool_kernel(const float* __restrict__ h, float* __restrict__ out) {
  __shared__ float s[256];
  int g = blockIdx.x, t = threadIdx.x;
  const float* base = h + (size_t)g * NPG * DD;
  float a = 0.f;
  for (int i = t; i < NPG * DD; i += 256) a += base[i];
  s[t] = a;
  __syncthreads();
  for (int o = 128; o > 0; o >>= 1) {
    if (t < o) s[t] += s[t + o];
    __syncthreads();
  }
  if (t == 0) out[g] = s[0] * (1.0f / (float)(NPG * DD));
}

// routing-balance loss scalar
__global__ void lbl_kernel(const float* __restrict__ easum, float* __restrict__ out) {
  float s = 0.f;
  for (int i = 0; i < LL * NE; i++) {
    float m = easum[i] * (1.0f / (float)EE);
    s += m * m;
  }
  out[NG] = s * ((float)NE / (float)LL) * 0.1f;  // * ROUTE_W
}

// ---------------- launch ----------------
extern "C" void kernel_launch(void* const* d_in, const int* in_sizes, int n_in,
                              void* d_out, int out_size, void* d_ws, size_t ws_size,
                              hipStream_t stream) {
  const float* pos   = (const float*)d_in[0];
  const float* emb   = (const float*)d_in[1];
  const float* ew1   = (const float*)d_in[2];
  const float* eb1   = (const float*)d_in[3];
  const float* ew2   = (const float*)d_in[4];
  const float* eb2   = (const float*)d_in[5];
  const float* nw1   = (const float*)d_in[6];
  const float* nb1   = (const float*)d_in[7];
  const float* nw2   = (const float*)d_in[8];
  const float* nb2   = (const float*)d_in[9];
  const float* gnw   = (const float*)d_in[10];
  const float* gnb   = (const float*)d_in[11];
  const float* gnms  = (const float*)d_in[12];
  const float* dnw   = (const float*)d_in[13];
  const float* dnb   = (const float*)d_in[14];
  const float* dnms  = (const float*)d_in[15];
  const int*   atype = (const int*)d_in[16];
  const int*   esrc  = (const int*)d_in[17];  // edge_index[0] = src

  char* ws = (char*)d_ws;
  float*     dist  = (float*)(ws + OFF_DIST);
  float*     zblk  = (float*)(ws + OFF_ZERO);
  float*     easum = zblk;          // 24
  float*     gsum  = zblk + 24;     // 32
  float*     gsum2 = zblk + 56;     // 32
  float*     gmean = (float*)(ws + OFF_GMEAN);
  float*     gistd = (float*)(ws + OFF_GISTD);
  float*     b2s   = (float*)(ws + OFF_B2SUM);
  float*     sumx  = (float*)(ws + OFF_SUMX);
  float*     sumx2 = (float*)(ws + OFF_SUMX2);
  float*     eaw   = (float*)(ws + OFF_EAW);
  _Float16*  agg   = (_Float16*)(ws + OFF_AGG);
  _Float16*  hA    = (_Float16*)(ws + OFF_HA);
  float*     hPre  = (float*)(ws + OFF_HPRE);
  _Float16*  w1t   = (_Float16*)(ws + OFF_W1T);
  _Float16*  w2t   = (_Float16*)(ws + OFF_W2T);
  float*     outp  = (float*)d_out;

  hipMemsetAsync(zblk, 0, 88 * sizeof(float), stream);

  init_h_kernel<<<(NN * DD) / 256, 256, 0, stream>>>(emb, atype, hA);
  dist_kernel<<<EE / 256, 256, 0, stream>>>(pos, esrc, dist, gsum, gsum2);
  gstat_kernel<<<1, 32, 0, stream>>>(gsum, gsum2, dnms, gmean, gistd);
  convert_w_kernel<<<(LL * NE * DD * HH) / 256, 256, 0, stream>>>(nw1, nw2, w1t, w2t);
  b2sum_kernel<<<3, 128, 0, stream>>>(nb2, b2s);

  for (int l = 0; l < LL; ++l) {
    router_kernel<<<EE / 256, 256, 0, stream>>>(dist, gmean, gistd,
                                                ew1 + l * EHID, eb1 + l * EHID,
                                                ew2 + l * EHID * NE, eb2 + l * NE,
                                                dnw, dnb, dnms, eaw, easum + l * NE);
    aggregate_kernel<<<NN / 4, 256, 0, stream>>>(hA, esrc, eaw, agg);
    moe_gemm_kernel<<<NN / 64, 256, 0, stream>>>(agg,
                                                 w1t + (size_t)l * NE * HH * DD,
                                                 w2t + (size_t)l * NE * DD * HH,
                                                 nb1 + (size_t)l * NE * HH,
                                                 b2s + l * DD, hPre);
    if (l + 1 < LL) {
      norm_reduce_kernel<<<NG, 256, 0, stream>>>(hPre, sumx, sumx2);
      norm_apply_kernel<<<(NN * DD) / 256, 256, 0, stream>>>(hPre, sumx, sumx2,
                                                             gnw + l * DD, gnb + l * DD,
                                                             gnms + l * DD, hA);
    }
  }
  pool_kernel<<<NG, 256, 0, stream>>>(hPre, outp);
  lbl_kernel<<<1, 1, 0, stream>>>(easum, outp);
}
